// CRM_14886356648008
// MI455X (gfx1250) — matrix operand order repaired and checked by reading, not compiled, not run on hardware
//
#include <hip/hip_runtime.h>
#include <hip/hip_bf16.h>

typedef __attribute__((ext_vector_type(16))) _Float16 v16h;
typedef __attribute__((ext_vector_type(8)))  float    v8f;
typedef __attribute__((ext_vector_type(4)))  unsigned int u32x4;
typedef __attribute__((ext_vector_type(8)))  int      i32x8;
typedef __attribute__((ext_vector_type(4)))  int      i32x4;

// Problem constants (from reference setup_inputs: B=4,C=32,H=W=128,scale=2)
constexpr int B_  = 4;
constexpr int C_  = 32;
constexpr int H_  = 128;
constexpr int W_  = 128;
constexpr int SC_ = 2;
constexpr int Ho_ = H_ * SC_;
constexpr int Wo_ = W_ * SC_;
constexpr int QB_ = Ho_ * Wo_;       // queries per batch
constexpr int TILE = 32;             // queries per workgroup (2 waves x 16)
constexpr int NTHREADS = 64;
constexpr int CH_STRIDE = H_ * W_;

// Pre-swizzled weight arena layout inside d_ws (element offsets, _Float16)
constexpr int WOFF0 = 0;                  // mlp_w0  : 64x256 (K padded from 36)
constexpr int WOFF1 = WOFF0 + 64 * 256;   // mlp_w1  : 256x256
constexpr int WOFF2 = WOFF1 + 256 * 256;  // mlp_w2  : 256x32
constexpr int WOFF3 = WOFF2 + 256 * 32;   // rout_w0 : 32x256
constexpr int WOFF4 = WOFF3 + 32 * 256;   // rout_w1 : 256x32
constexpr int WOFF5 = WOFF4 + 256 * 32;   // off_w0  : 32x256
constexpr int WOFF6 = WOFF5 + 32 * 256;   // off_w1  : 256x16 (N padded from 2)
constexpr int WTOT  = WOFF6 + 256 * 16;   // = 118784 halfs (237568 B, 256B-aligned)
// Padded biases (float) appended after weights
constexpr int BOFF0 = 0, BOFF1 = 256, BOFF2 = 512, BOFF3 = 544,
              BOFF4 = 800, BOFF5 = 832, BOFF6 = 1088;

#if __has_builtin(__builtin_amdgcn_tensor_load_to_lds) && \
    __has_builtin(__builtin_amdgcn_s_wait_tensorcnt)
#define LIIF_TDM 1
#endif

// ---------- fragment-order indexing (CDNA5 16x16x32 f16 striping) ----------
// A (16x32 MxK): lanes 0-15 hold K {0..7,16..23}; lanes 16-31 hold K {8..15,24..31}.
__device__ __forceinline__ int a_idx(int q, int k, int Ktot) {
  int qt = q >> 4, m = q & 15;
  int kg = k >> 5, kk = k & 31;
  int sel = (kk >> 3) & 1;
  int i = (kk & 7) + ((kk >> 4) << 3);
  int lane = m + (sel << 4);
  return ((qt * (Ktot >> 5) + kg) * 32 + lane) * 16 + i;
}
// B (32x16 KxN) within a 32-row chunk: lanes 0-15 (N=lane) hold K {0..7,16..23},
// lanes 16-31 hold K {8..15,24..31}.
__device__ __forceinline__ int b_idx(int r, int c) {
  int nt = c >> 4;
  int lane = (c & 15) + (((r >> 3) & 1) << 4);
  return (nt * 32 + lane) * 16 + ((r & 7) + ((r >> 4) << 3));
}
__device__ __forceinline__ v16h ld_frag(const _Float16* p) {
  return *(const v16h*)p;                  // 32B-aligned contiguous fragment
}

// ---------- weight-chunk staging: TDM DMA (async tensor) or b128 copy -------
__device__ __forceinline__ void stage_weights(_Float16* dst, const _Float16* src,
                                              int nhalf, int tid) {
#if defined(LIIF_TDM)
  if (tid < 32) {  // one wave issues the DMA; EXEC-uniform branch
    unsigned ldsOff = (unsigned)(size_t)(__attribute__((address_space(3))) _Float16*)dst;
    unsigned long long ga = (unsigned long long)(size_t)src;
    // D# group0: count=1 | lds_addr | global_addr[56:0] | type=2
    u32x4 g0;
    g0[0] = 1u;
    g0[1] = ldsOff;
    g0[2] = (unsigned)ga;
    g0[3] = ((unsigned)(ga >> 32) & 0x01FFFFFFu) | 0x80000000u;
    // D# group1: data_size=2B; 1 row of nhalf elements; tile == tensor
    i32x8 g1;
    g1[0] = 0x00010000;                        // workgroup_mask=0, data_size=1 (2B)
    g1[1] = (int)((nhalf & 0xFFFF) << 16);     // tensor_dim0[15:0] at bits 63:48
    g1[2] = (int)(((unsigned)nhalf >> 16) | (1u << 16)); // dim0[31:16], tensor_dim1=1
    g1[3] = (int)((unsigned)nhalf << 16);      // tile_dim0 at bits 127:112
    g1[4] = 1;                                 // tile_dim1=1, tile_dim2=0
    g1[5] = nhalf;                             // tensor_dim0_stride[31:0]
    g1[6] = 0;
    g1[7] = 0;
    i32x4 z4 = {0, 0, 0, 0};
#if __clang_major__ >= 23
    i32x8 z8 = {0, 0, 0, 0, 0, 0, 0, 0};
    __builtin_amdgcn_tensor_load_to_lds(g0, g1, z4, z4, z8, 0);
#else
    __builtin_amdgcn_tensor_load_to_lds(g0, g1, z4, z4, 0);
#endif
    __builtin_amdgcn_s_wait_tensorcnt(0);
  }
#else
  for (int i = tid * 8; i < nhalf; i += NTHREADS * 8)
    *(uint4*)(dst + i) = *(const uint4*)(src + i);
#endif
}

// ---------- prep pass: f32 weights -> f16, padded, B-fragment chunk order ----
template <int K, int NP>
__global__ void __launch_bounds__(256)
prep_layer(const float* __restrict__ gw, int Krows, int Ncols,
           const float* __restrict__ gb,
           _Float16* __restrict__ dw, float* __restrict__ db) {
  int i = blockIdx.x * 256 + threadIdx.x;
  if (i < K * NP) {
    int kr = i / NP, c = i % NP;
    int kc = kr >> 5, r = kr & 31;
    float v = (kr < Krows && c < Ncols) ? gw[(size_t)kr * Ncols + c] : 0.0f;
    dw[kc * 32 * NP + b_idx(r, c)] = (_Float16)v;
  }
  if (i < NP) db[i] = (i < Ncols) ? gb[i] : 0.0f;
}

// ---------- bicubic helpers (torch aten A=-0.75 formulation) ----------
__device__ __forceinline__ void cubw(float t, float w[4]) {
  const float A = -0.75f;
  float x = t + 1.0f;
  w[0] = ((A * x - 5.0f * A) * x + 8.0f * A) * x - 4.0f * A;
  w[1] = ((A + 2.0f) * t - (A + 3.0f)) * t * t + 1.0f;
  float s = 1.0f - t;
  w[2] = ((A + 2.0f) * s - (A + 3.0f)) * s * s + 1.0f;
  w[3] = 1.0f - w[0] - w[1] - w[2];
}

__device__ __forceinline__ void bicubic16(const float* __restrict__ xb,
                                          float gx, float gy, float acc[16],
                                          float* qcy, float* qcx, bool wantc) {
  float ix = ((gx + 1.0f) * (float)W_ - 1.0f) * 0.5f;
  float iy = ((gy + 1.0f) * (float)H_ - 1.0f) * 0.5f;
  float fx0 = floorf(ix), fy0 = floorf(iy);
  int ix0 = (int)fx0, iy0 = (int)fy0;
  float wx[4], wy[4];
  cubw(ix - fx0, wx);
  cubw(iy - fy0, wy);
#pragma unroll
  for (int c = 0; c < 16; ++c) acc[c] = 0.0f;
  float cyv = 0.0f, cxv = 0.0f;
#pragma unroll
  for (int i = 0; i < 4; ++i) {
    int yi = iy0 - 1 + i;
    bool yok = (yi >= 0) && (yi < H_);
    int yc = yi < 0 ? 0 : (yi > H_ - 1 ? H_ - 1 : yi);
#pragma unroll
    for (int j = 0; j < 4; ++j) {
      int xj = ix0 - 1 + j;
      bool ok = yok && (xj >= 0) && (xj < W_);
      int xc = xj < 0 ? 0 : (xj > W_ - 1 ? W_ - 1 : xj);
      float wgt = wy[i] * wx[j] * (ok ? 1.0f : 0.0f);
      const float* p = xb + yc * W_ + xc;
#pragma unroll
      for (int c = 0; c < 16; ++c) acc[c] = fmaf(wgt, p[c * CH_STRIDE], acc[c]);
      if (wantc) {
        cyv = fmaf(wgt, -1.0f + (2.0f * yc + 1.0f) / (float)H_, cyv);
        cxv = fmaf(wgt, -1.0f + (2.0f * xc + 1.0f) / (float)W_, cxv);
      }
    }
  }
  if (wantc) { *qcy = cyv; *qcx = cxv; }
}

// ---------- one dense layer: [16 q x K] @ [K x NP] + bias (+relu) ----------
// Weights arrive pre-swizzled f16 -> staging is a TDM DMA (or b128 copy).
// First k-step uses inline-zero C; bias folded into the epilogue. B fragments
// are software-pipelined (load nt+1 while WMMA nt executes).
template <int K, int NP, bool RELU, bool OF32>
__device__ __forceinline__ void layer_gemm(const _Float16* Act,
                                           const _Float16* __restrict__ gws,
                                           const float* __restrict__ gbs,
                                           _Float16* outH, float* outF,
                                           _Float16* Wb, float* BiasS,
                                           int tid, int lane, int qt) {
  constexpr int NT = NP / 16;
  constexpr int KC = K / 32;
  constexpr int CH = 32 * NP;              // halfs per staged chunk
  __syncthreads();  // previous phase's LDS fully consumed/produced
  for (int i = tid; i < NP; i += NTHREADS) BiasS[i] = gbs[i];
  stage_weights(Wb, gws, CH, tid);
  __syncthreads();
  if (KC > 1) __builtin_prefetch(gws + CH, 0, 1);
  int n0l = lane & 15;
  v8f acc[NT];
  {
    v16h a = ld_frag(Act + ((qt * KC) * 32 + lane) * 16);
    v16h bc = ld_frag(Wb + lane * 16);
    v8f z = {};
#pragma unroll
    for (int nt = 0; nt < NT; ++nt) {
      v16h bn = (nt + 1 < NT) ? ld_frag(Wb + ((nt + 1) * 32 + lane) * 16) : bc;
      acc[nt] = __builtin_amdgcn_wmma_f32_16x16x32_f16(false, a, false, bc,
                                                       (short)0, z, false, false);
      bc = bn;
    }
  }
  for (int kc = 1; kc < KC; ++kc) {
    __syncthreads();
    stage_weights(Wb, gws + kc * CH, CH, tid);
    __syncthreads();
    if (kc + 1 < KC) __builtin_prefetch(gws + (kc + 1) * CH, 0, 1);
    v16h a = ld_frag(Act + ((qt * KC + kc) * 32 + lane) * 16);
    v16h bc = ld_frag(Wb + lane * 16);
#pragma unroll
    for (int nt = 0; nt < NT; ++nt) {
      v16h bn = (nt + 1 < NT) ? ld_frag(Wb + ((nt + 1) * 32 + lane) * 16) : bc;
      acc[nt] = __builtin_amdgcn_wmma_f32_16x16x32_f16(false, a, false, bc,
                                                       (short)0, acc[nt], false, false);
      bc = bn;
    }
  }
  // epilogue: element (m = r + 8*(lane>=16), n = lane&15); bias + (relu) here
  int mb = (lane >> 4) * 8;
#pragma unroll
  for (int nt = 0; nt < NT; ++nt) {
    float bv = BiasS[nt * 16 + n0l];
#pragma unroll
    for (int r = 0; r < 8; ++r) {
      float v = acc[nt][r] + bv;
      if (RELU) v = fmaxf(v, 0.0f);
      int qrow = qt * 16 + mb + r;
      int col = nt * 16 + n0l;
      if constexpr (OF32) outF[qrow * NP + col] = v;
      else                outH[a_idx(qrow, col, NP)] = (_Float16)v;
    }
  }
}

// ---------- fused LIIF forward kernel ----------
__global__ void __launch_bounds__(NTHREADS)
liif_fused(const float* __restrict__ x,
           const _Float16* __restrict__ Wall, const float* __restrict__ Ball,
           float* __restrict__ out) {
  __shared__ __align__(32) _Float16 sAin[TILE * 64];   // A-fragment order, K=64
  __shared__ __align__(32) _Float16 sWb[32 * 256];     // B-fragment order chunk
  __shared__ __align__(32) _Float16 sH1[TILE * 256];   // A-fragment order
  __shared__ __align__(32) _Float16 sH2[TILE * 256];   // A-fragment order
  __shared__ __align__(32) _Float16 sPred[TILE * 32];  // A-fragment order
  __shared__ float sBias[256];
  __shared__ float sRout[TILE * 32];
  __shared__ float sOff[TILE * 16];
  __shared__ float sGx[TILE], sGy[TILE], sGx2[TILE], sGy2[TILE];

  const int tid = threadIdx.x;
  const int lane = tid & 31;
  const int qt = tid >> 5;                 // wave's 16-query M-tile index

  // ---- phase 1: bicubic sampling + input assembly ----
  const int q = tid >> 1;                  // 0..31 local query
  const int half = tid & 1;                // channel half
  const int qg = blockIdx.x * TILE + q;
  const int b = qg / QB_;
  const int rem = qg % QB_;
  const int ho = rem / Wo_, wo = rem % Wo_;
  const float cy = -1.0f + (2.0f * ho + 1.0f) / (float)Ho_;
  const float cx = -1.0f + (2.0f * wo + 1.0f) / (float)Wo_;
  const float eps = 1e-6f;
  const float gy = fminf(fmaxf(cy, -1.0f + eps), 1.0f - eps);
  const float gx = fminf(fmaxf(cx, -1.0f + eps), 1.0f - eps);
  const float* xb = x + ((size_t)b * C_ + half * 16) * CH_STRIDE;
  {
    float acc[16], qcy = 0.0f, qcx = 0.0f;
    bicubic16(xb, gx, gy, acc, &qcy, &qcx, half == 0);
#pragma unroll
    for (int c = 0; c < 16; ++c)
      sAin[a_idx(q, half * 16 + c, 64)] = (_Float16)acc[c];
    if (half == 0) {
      float rel0 = (cy - qcy) * (float)H_;
      float rel1 = (cx - qcx) * (float)W_;
      // faithful to the reference's cell.at[:,0]/.at[:,1] (query-dim) quirk
      float f = (rem == 0) ? 2.0f / (float)Ho_ : ((rem == 1) ? 2.0f / (float)Wo_ : 1.0f);
      sAin[a_idx(q, 32, 64)] = (_Float16)rel0;
      sAin[a_idx(q, 33, 64)] = (_Float16)rel1;
      sAin[a_idx(q, 34, 64)] = (_Float16)(f * (float)H_);
      sAin[a_idx(q, 35, 64)] = (_Float16)(f * (float)W_);
      sGx[q] = gx; sGy[q] = gy;
    } else {
#pragma unroll
      for (int k2 = 36; k2 < 64; ++k2) sAin[a_idx(q, k2, 64)] = (_Float16)0.0f;
    }
  }

  // ---- MLP stack (all WMMA) ----
  layer_gemm<64, 256, true,  false>(sAin,  Wall + WOFF0, Ball + BOFF0, sH1,  nullptr, sWb, sBias, tid, lane, qt);
  layer_gemm<256,256, true,  false>(sH1,   Wall + WOFF1, Ball + BOFF1, sH2,  nullptr, sWb, sBias, tid, lane, qt);
  layer_gemm<256, 32, false, false>(sH2,   Wall + WOFF2, Ball + BOFF2, sPred,nullptr, sWb, sBias, tid, lane, qt);
  layer_gemm<32, 256, true,  false>(sPred, Wall + WOFF3, Ball + BOFF3, sH1,  nullptr, sWb, sBias, tid, lane, qt);
  layer_gemm<256, 32, false, true >(sH1,   Wall + WOFF4, Ball + BOFF4, nullptr, sRout, sWb, sBias, tid, lane, qt);
  layer_gemm<32, 256, true,  false>(sPred, Wall + WOFF5, Ball + BOFF5, sH2,  nullptr, sWb, sBias, tid, lane, qt);
  layer_gemm<256, 16, false, true >(sH2,   Wall + WOFF6, Ball + BOFF6, nullptr, sOff,  sWb, sBias, tid, lane, qt);

  __syncthreads();
  if (tid < TILE) {
    sGx2[tid] = sGx[tid] + sOff[tid * 16 + 0];
    sGy2[tid] = sGy[tid] + sOff[tid * 16 + 1];
  }
  __syncthreads();

  // ---- phase 3: offset bicubic resample + routing modulation ----
  {
    float acc[16];
    bicubic16(xb, sGx2[q], sGy2[q], acc, nullptr, nullptr, false);
    size_t obase = ((size_t)b * C_ + half * 16) * (size_t)(Ho_ * Wo_) +
                   (size_t)ho * Wo_ + wo;
#pragma unroll
    for (int c = 0; c < 16; ++c) {
      float r = sRout[q * 32 + half * 16 + c];
      out[obase + (size_t)c * (Ho_ * Wo_)] = acc[c] * (1.0f + r);
    }
  }
}

extern "C" void kernel_launch(void* const* d_in, const int* in_sizes, int n_in,
                              void* d_out, int out_size, void* d_ws, size_t ws_size,
                              hipStream_t stream) {
  (void)in_sizes; (void)n_in; (void)out_size; (void)ws_size;
  const float* x   = (const float*)d_in[0];
  const float* w0  = (const float*)d_in[1];
  const float* b0  = (const float*)d_in[2];
  const float* w1  = (const float*)d_in[3];
  const float* b1  = (const float*)d_in[4];
  const float* w2  = (const float*)d_in[5];
  const float* b2  = (const float*)d_in[6];
  const float* rw0 = (const float*)d_in[7];
  const float* rb0 = (const float*)d_in[8];
  const float* rw1 = (const float*)d_in[9];
  const float* rb1 = (const float*)d_in[10];
  const float* ow0 = (const float*)d_in[11];
  const float* ob0 = (const float*)d_in[12];
  const float* ow1 = (const float*)d_in[13];
  const float* ob1 = (const float*)d_in[14];
  float* out = (float*)d_out;

  _Float16* Wall = (_Float16*)d_ws;                       // 237568 B
  float*    Ball = (float*)((char*)d_ws + WTOT * 2);      // 256B-aligned tail

  // ---- prep pass: pad + swizzle + f16-convert all weights/biases ----
  prep_layer<64, 256><<<(64 * 256 + 255) / 256, 256, 0, stream>>>(w0, 36, 256, b0, Wall + WOFF0, Ball + BOFF0);
  prep_layer<256,256><<<(256 * 256 + 255) / 256, 256, 0, stream>>>(w1, 256, 256, b1, Wall + WOFF1, Ball + BOFF1);
  prep_layer<256, 32><<<(256 * 32 + 255) / 256, 256, 0, stream>>>(w2, 256, 32, b2, Wall + WOFF2, Ball + BOFF2);
  prep_layer<32, 256><<<(32 * 256 + 255) / 256, 256, 0, stream>>>(rw0, 32, 256, rb0, Wall + WOFF3, Ball + BOFF3);
  prep_layer<256, 32><<<(256 * 32 + 255) / 256, 256, 0, stream>>>(rw1, 256, 32, rb1, Wall + WOFF4, Ball + BOFF4);
  prep_layer<32, 256><<<(32 * 256 + 255) / 256, 256, 0, stream>>>(ow0, 32, 256, ob0, Wall + WOFF5, Ball + BOFF5);
  prep_layer<256, 16><<<(256 * 16 + 255) / 256, 256, 0, stream>>>(ow1, 256, 2, ob1, Wall + WOFF6, Ball + BOFF6);

  // ---- fused main pass ----
  const int nquery = B_ * QB_;             // 262144
  const int blocks = nquery / TILE;        // 8192
  liif_fused<<<blocks, NTHREADS, 0, stream>>>(x, Wall, Ball, out);
}
